// STConvBlockParallelGAT_60490319397017
// MI455X (gfx1250) — matrix operand order repaired and checked by reading, not compiled, not run on hardware
//
#include <hip/hip_runtime.h>
#include <hip/hip_bf16.h>
#include <math.h>

#define BB 16
#define CIN 2
#define T0 24
#define NN 400
#define NP 416
#define CC 64
#define T1 22
#define T2 20
#define BT1 (BB*T1)   /* 352 */
#define BT2 (BB*T2)   /* 320 */
#define HH 2
#define DD 32
#define NEGV -1000000000.0f
#define EPSV 1e-5f

typedef _Float16 f16;
typedef __attribute__((ext_vector_type(16))) _Float16 v16h;
typedef __attribute__((ext_vector_type(8)))  float    v8f;

// ---- WMMA fragment loaders (contiguous k order, wave32) -------------
// A (16x32 f16), row-major [m][k]: lane m = lane&15, hi = lane>>4;
// elem e -> k = 16*hi + e (contiguous, matches ldBT). Two b128 runs.
__device__ __forceinline__ v16h ldA(const f16* A, int lda, int lane) {
  int m = lane & 15, hi = lane >> 4;
  const f16* r = A + (long)m * lda + 16 * hi;
  v16h a;
#pragma unroll
  for (int e = 0; e < 8; ++e) a[e] = r[e];
#pragma unroll
  for (int e = 0; e < 8; ++e) a[8 + e] = r[8 + e];
  return a;
}
// B (32x16 f16) from TRANSPOSED storage [n][k] (k contiguous):
// lane n = lane&15, hi = lane>>4; elem e -> k = 16*hi + e.
// 32 contiguous bytes per lane -> two b128 loads.
__device__ __forceinline__ v16h ldBT(const f16* BT, int ldk, int lane) {
  int n = lane & 15, hi = lane >> 4;
  const f16* p = BT + (long)n * ldk + 16 * hi;
  v16h b;
#pragma unroll
  for (int e = 0; e < 16; ++e) b[e] = p[e];
  return b;
}

#define WMMA(acc, a, b) \
  acc = __builtin_amdgcn_wmma_f32_16x16x32_f16(false, a, false, b, (short)0, acc, false, false)

__device__ __forceinline__ float sigm(float x) { return 1.0f / (1.0f + expf(-x)); }

// ---------------------------------------------------------------------------
// prep: f16 weight/gso staging (B-side transposed), folded attention vectors,
// pad zeroing for the transposed (K-side) node buffers
// ---------------------------------------------------------------------------
__global__ void k_prep(const float* gso, const float* wq, const float* wk,
                       const float* wv, const float* a_src, const float* a_dst,
                       const float* cheb_w, const float* conv2_w,
                       f16* Gf, f16* wvh, f16* chWt, f16* c2Wt,
                       float* aq, float* ak,
                       f16* x1t, f16* y1t, f16* vt) {
  const long NG = (long)NN * NP;      // gso f16 A-matrix, K-padded
  const long NW = CC * CC;            // wv as-is: already [co][ci] = B^T
  const long NC = 3 * CC * CC;        // cheb W transposed [k][co][ci]
  const long N2 = 3 * 128 * CC;       // conv2 W transposed [kt][cp][c]
  const long NA = CC * HH;            // folded a_src / a_dst
  const long NZ = (long)BT1 * CC * 16;// pad cols per transposed buffer
  const long total = NG + NW + NC + N2 + NA + NA + 3 * NZ;
  for (long idx = blockIdx.x * (long)blockDim.x + threadIdx.x; idx < total;
       idx += (long)gridDim.x * blockDim.x) {
    long i = idx;
    if (i < NG) { long h = i / NP, c = i % NP;
      Gf[i] = (c < NN) ? (f16)gso[h * NN + c] : (f16)0.f; continue; }
    i -= NG;
    if (i < NW) { wvh[i] = (f16)wv[i]; continue; }
    i -= NW;
    if (i < NC) { long k = i / (CC * CC); long r = i % (CC * CC);
      long co = r / CC, ci = r % CC;
      chWt[i] = (f16)cheb_w[(k * CC + ci) * CC + co]; continue; }
    i -= NC;
    if (i < N2) { long kt = i / (128 * CC); long r = i % (128 * CC);
      long cp = r / CC, c = r % CC;
      c2Wt[i] = (f16)conv2_w[(cp * CC + c) * 3 + kt]; continue; }
    i -= N2;
    if (i < NA) { long ci = i / HH, h = i % HH; float s = 0.f;
      for (int d = 0; d < DD; ++d) s += wq[(h * DD + d) * CC + ci] * a_src[h * DD + d];
      aq[i] = s; continue; }
    i -= NA;
    if (i < NA) { long ci = i / HH, h = i % HH; float s = 0.f;
      for (int d = 0; d < DD; ++d) s += wk[(h * DD + d) * CC + ci] * a_dst[h * DD + d];
      ak[i] = s; continue; }
    i -= NA;
    long buf = i / NZ; long r = i % NZ;
    long s = r / (CC * 16); long rem = r % (CC * 16);
    long c = rem / 16, pr = rem % 16;
    long off = (s * CC + c) * NP + NN + pr;
    f16* dst = (buf == 0) ? x1t : (buf == 1) ? y1t : vt;
    dst[off] = (f16)0.f;
  }
}

// ---------------------------------------------------------------------------
// tconv1 + GLU (C_IN=2, tiny -> scalar).
// x1f: f32 [s][n][c]; x1h: f16 [s][n][c] (A-side); x1t: f16 [s][c][n] (B-side)
// ---------------------------------------------------------------------------
__global__ void k_tconv1(const float* x, const float* w, const float* bias,
                         float* x1f, f16* x1h, f16* x1t) {
  long idx = blockIdx.x * (long)blockDim.x + threadIdx.x;
  const long total = (long)BT1 * NN * CC;
  if (idx >= total) return;
  int c = idx % CC; int n = (idx / CC) % NN; int s = idx / ((long)CC * NN);
  int b = s / T1, t = s % T1;
  float p = bias[c], q = bias[CC + c];
#pragma unroll
  for (int kt = 0; kt < 3; ++kt) {
    float xv0 = x[(((long)b * CIN + 0) * T0 + t + kt) * NN + n];
    float xv1 = x[(((long)b * CIN + 1) * T0 + t + kt) * NN + n];
    p += xv0 * w[(c * CIN + 0) * 3 + kt]        + xv1 * w[(c * CIN + 1) * 3 + kt];
    q += xv0 * w[((CC + c) * CIN + 0) * 3 + kt] + xv1 * w[((CC + c) * CIN + 1) * 3 + kt];
  }
  float xin = (c < CIN) ? x[(((long)b * CIN + c) * T0 + t + 2) * NN + n] : 0.f;
  float v = (p + xin) * sigm(q);
  x1f[((long)s * NN + n) * CC + c] = v;
  x1h[((long)s * NN + n) * CC + c] = (f16)v;
  x1t[((long)s * CC + c) * NP + n] = (f16)v;
}

// ---------------------------------------------------------------------------
// y1 = G @ x1 per slice. M=400 (25 tiles), N=64 (4 waves), K=416 (13 tiles)
// writes y1h [s][n][c] (A-side for cheb) and y1t [s][c][n] (B-side for G@y1)
// ---------------------------------------------------------------------------
__global__ void __launch_bounds__(128) k_gx(const f16* Gf, const f16* x1t,
                                            f16* y1h, f16* y1t) {
  int blk = blockIdx.x; int mt = blk % 25; int s = blk / 25;
  int wv = threadIdx.x >> 5, lane = threadIdx.x & 31;
  int m0 = mt * 16, c0 = wv * 16;
  const f16* Ab = Gf + (long)m0 * NP;
  const f16* Bb = x1t + ((long)s * CC + c0) * NP;
  v8f acc = {};
  for (int kt = 0; kt < 13; ++kt) {
    v16h a = ldA(Ab + kt * 32, NP, lane);
    v16h b = ldBT(Bb + kt * 32, NP, lane);
    WMMA(acc, a, b);
  }
  int n = lane & 15, hi = lane >> 4;
#pragma unroll
  for (int r = 0; r < 8; ++r) {
    int m = 8 * hi + r;
    f16 v = (f16)acc[r];
    y1h[((long)s * NN + m0 + m) * CC + c0 + n] = v;
    y1t[((long)s * CC + c0 + n) * NP + m0 + m] = v;
  }
}

// y2 = 2*(G @ y1) - x1   (A-side output only)
__global__ void __launch_bounds__(128) k_gy(const f16* Gf, const f16* y1t,
                                            const float* x1f, f16* y2h) {
  int blk = blockIdx.x; int mt = blk % 25; int s = blk / 25;
  int wv = threadIdx.x >> 5, lane = threadIdx.x & 31;
  int m0 = mt * 16, c0 = wv * 16;
  const f16* Ab = Gf + (long)m0 * NP;
  const f16* Bb = y1t + ((long)s * CC + c0) * NP;
  v8f acc = {};
  for (int kt = 0; kt < 13; ++kt) {
    v16h a = ldA(Ab + kt * 32, NP, lane);
    v16h b = ldBT(Bb + kt * 32, NP, lane);
    WMMA(acc, a, b);
  }
  int n = lane & 15, hi = lane >> 4;
#pragma unroll
  for (int r = 0; r < 8; ++r) {
    int m = 8 * hi + r;
    float v = 2.f * acc[r] - x1f[((long)s * NN + m0 + m) * CC + c0 + n];
    y2h[((long)s * NN + m0 + m) * CC + c0 + n] = (f16)v;
  }
}

// V = X @ wv^T  (K=64 -> 2 tiles); output transposed [s][c][node] (B-side)
__global__ void __launch_bounds__(128) k_v(const f16* x1h, const f16* wvh, f16* vt) {
  int blk = blockIdx.x; int mt = blk % 25; int s = blk / 25;
  int wv = threadIdx.x >> 5, lane = threadIdx.x & 31;
  int m0 = mt * 16, c0 = wv * 16;
  const f16* Ab = x1h + ((long)s * NN + m0) * CC;
  const f16* Bb = wvh + (long)c0 * CC;   // [co][ci], k=ci contiguous
  v8f acc = {};
#pragma unroll
  for (int k2 = 0; k2 < 2; ++k2) {
    v16h a = ldA(Ab + k2 * 32, CC, lane);
    v16h b = ldBT(Bb + k2 * 32, CC, lane);
    WMMA(acc, a, b);
  }
  int n = lane & 15, hi = lane >> 4;
#pragma unroll
  for (int r = 0; r < 8; ++r) {
    int m = 8 * hi + r;
    vt[((long)s * CC + c0 + n) * NP + m0 + m] = (f16)acc[r];
  }
}

// s_src/s_dst via folded vectors: s_src[n,h] = X[n,:] . aq[:,h]
__global__ void k_ssd(const float* x1f, const float* aq, const float* ak,
                      float* ssrc, float* sdst) {
  long idx = blockIdx.x * (long)blockDim.x + threadIdx.x;
  const long total = (long)BT1 * NN * HH;
  if (idx >= total) return;
  int h = idx % HH; long sn = idx / HH;
  const float* xr = x1f + sn * CC;
  float s1 = 0.f, s2 = 0.f;
#pragma unroll 8
  for (int ci = 0; ci < CC; ++ci) { s1 += xr[ci] * aq[ci * HH + h]; s2 += xr[ci] * ak[ci * HH + h]; }
  ssrc[idx] = s1; sdst[idx] = s2;
}

// x_gc = sum_k Xk @ Wk + cheb_b + x1   (3 mats x 2 K-tiles accumulated)
__global__ void __launch_bounds__(128) k_cheb(const f16* x1h, const f16* y1h,
                                              const f16* y2h, const f16* chWt,
                                              const float* cheb_b, const float* x1f,
                                              float* gc) {
  int blk = blockIdx.x; int mt = blk % 25; int s = blk / 25;
  int wv = threadIdx.x >> 5, lane = threadIdx.x & 31;
  int m0 = mt * 16, c0 = wv * 16;
  const f16* mats[3] = { x1h, y1h, y2h };
  v8f acc = {};
#pragma unroll
  for (int k = 0; k < 3; ++k) {
    const f16* Ab = mats[k] + ((long)s * NN + m0) * CC;
    const f16* Bb = chWt + ((long)k * CC + c0) * CC;   // [k][co][ci]
#pragma unroll
    for (int k2 = 0; k2 < 2; ++k2) {
      v16h a = ldA(Ab + k2 * 32, CC, lane);
      v16h b = ldBT(Bb + k2 * 32, CC, lane);
      WMMA(acc, a, b);
    }
  }
  int n = lane & 15, hi = lane >> 4;
#pragma unroll
  for (int r = 0; r < 8; ++r) {
    int m = 8 * hi + r;
    long o = ((long)s * NN + m0 + m) * CC + c0 + n;
    gc[o] = acc[r] + cheb_b[c0 + n] + x1f[o];
  }
}

// ---------------------------------------------------------------------------
// GAT attention: per (slice, head, 16-row tile): scores->softmax in LDS,
// un-normalized exp tile through WMMA vs V^T, row-rescale after.
// ---------------------------------------------------------------------------
__global__ void __launch_bounds__(32) k_attn(const float* gso, const float* ssrc,
                                             const float* sdst, const f16* vt,
                                             float* gat) {
  __shared__ f16 P[16 * NP];
  __shared__ float rinv[16];
  int blk = blockIdx.x;
  int mt = blk % 25; int h = (blk / 25) & 1; int s = blk / 50;
  int lane = threadIdx.x;
  int m0 = mt * 16;
  if (lane < 16) {
    int i = m0 + lane;
    float si = ssrc[((long)s * NN + i) * HH + h];
    const float* gr = gso + (long)i * NN;
    const float* sd = sdst + (long)s * NN * HH + h;
    float mx = -3.0e38f;
    for (int j = 0; j < NN; ++j) {
      float sc = si + sd[(long)j * HH];
      sc = (sc >= 0.f) ? sc : 0.2f * sc;
      if (gr[j] == 0.f) sc += NEGV;
      mx = fmaxf(mx, sc);
    }
    float sum = 0.f;
    for (int j = 0; j < NN; ++j) {
      float sc = si + sd[(long)j * HH];
      sc = (sc >= 0.f) ? sc : 0.2f * sc;
      if (gr[j] == 0.f) sc += NEGV;
      float e = expf(sc - mx);
      sum += e;
      P[lane * NP + j] = (f16)e;
    }
    for (int j = NN; j < NP; ++j) P[lane * NP + j] = (f16)0.f;
    rinv[lane] = 1.0f / sum;
  }
  __syncthreads();
#pragma unroll
  for (int d0 = 0; d0 < 2; ++d0) {
    const f16* Vb = vt + ((long)s * CC + h * DD + d0 * 16) * NP;  // [c][node]
    v8f acc = {};
    for (int kt = 0; kt < 13; ++kt) {
      v16h a = ldA(&P[kt * 32], NP, lane);
      v16h b = ldBT(Vb + kt * 32, NP, lane);
      WMMA(acc, a, b);
    }
    int n = lane & 15, hi = lane >> 4;
#pragma unroll
    for (int r = 0; r < 8; ++r) {
      int m = 8 * hi + r;
      gat[((long)s * NN + m0 + m) * CC + h * DD + d0 * 16 + n] = acc[r] * rinv[m];
    }
  }
}

// GAT LayerNorm(C) + combine: x_f = relu(x_gc + LN(gat))
__global__ void k_lncomb(const float* gat, const float* gc, const float* g,
                         const float* beta, float* xf, f16* xfh) {
  long idx = blockIdx.x * (long)blockDim.x + threadIdx.x;
  const long total = (long)BT1 * NN;
  if (idx >= total) return;
  const float* v = gat + idx * CC;
  float mu = 0.f;
#pragma unroll 8
  for (int c = 0; c < CC; ++c) mu += v[c];
  mu *= (1.0f / CC);
  float var = 0.f;
#pragma unroll 8
  for (int c = 0; c < CC; ++c) { float d = v[c] - mu; var += d * d; }
  var *= (1.0f / CC);
  float rs = rsqrtf(var + EPSV);
#pragma unroll 8
  for (int c = 0; c < CC; ++c) {
    float ln = (v[c] - mu) * rs * g[c] + beta[c];
    float o = gc[idx * CC + c] + ln;
    o = fmaxf(o, 0.f);
    xf[idx * CC + c] = o;
    xfh[idx * CC + c] = (f16)o;
  }
}

// ---------------------------------------------------------------------------
// tconv2 + GLU via WMMA: out = sum_kt X(t+kt) @ W2[kt]; per wave p & q tiles
// ---------------------------------------------------------------------------
__global__ void __launch_bounds__(128) k_tconv2(const f16* xfh, const float* xf,
                                                const f16* c2Wt, const float* bias2,
                                                float* x2) {
  int blk = blockIdx.x; int mt = blk % 25; int s2 = blk / 25;
  int b = s2 / T2, t2 = s2 % T2;
  int wv = threadIdx.x >> 5, lane = threadIdx.x & 31;
  int m0 = mt * 16;
  int cp0 = wv * 16, cq0 = 64 + wv * 16;
  v8f accp = {}, accq = {};
#pragma unroll
  for (int kt = 0; kt < 3; ++kt) {
    long sA = (long)(b * T1 + t2 + kt) * NN;
    const f16* Ab = xfh + (sA + m0) * CC;
    const f16* Bk = c2Wt + (long)kt * 128 * CC;   // [kt][cp][c]
#pragma unroll
    for (int k2 = 0; k2 < 2; ++k2) {
      v16h a  = ldA(Ab + k2 * 32, CC, lane);
      v16h bp = ldBT(Bk + (long)cp0 * CC + k2 * 32, CC, lane);
      WMMA(accp, a, bp);
      v16h bq = ldBT(Bk + (long)cq0 * CC + k2 * 32, CC, lane);
      WMMA(accq, a, bq);
    }
  }
  int n = lane & 15, hi = lane >> 4;
  long sin = (long)(b * T1 + t2 + 2) * NN;
#pragma unroll
  for (int r = 0; r < 8; ++r) {
    int m = 8 * hi + r; int node = m0 + m;
    float p = accp[r] + bias2[cp0 + n];
    float q = accq[r] + bias2[cq0 + n];
    float xin = xf[(sin + node) * CC + cp0 + n];
    x2[((long)s2 * NN + node) * CC + cp0 + n] = (p + xin) * sigm(q);
  }
}

// Final LayerNorm over (N,C) per (b,t2); write (B,C,T2,N)
__global__ void __launch_bounds__(256) k_finln(const float* x2, const float* g,
                                               const float* beta, float* out) {
  __shared__ float ss[256], sq[256];
  int s2 = blockIdx.x; int tid = threadIdx.x;
  int b = s2 / T2, t2 = s2 % T2;
  const float* xp = x2 + (long)s2 * NN * CC;
  float s = 0.f, q = 0.f;
  for (int i = tid; i < NN * CC; i += 256) { float v = xp[i]; s += v; q += v * v; }
  ss[tid] = s; sq[tid] = q; __syncthreads();
  for (int st = 128; st > 0; st >>= 1) {
    if (tid < st) { ss[tid] += ss[tid + st]; sq[tid] += sq[tid + st]; }
    __syncthreads();
  }
  float mu = ss[0] * (1.0f / (NN * CC));
  float var = sq[0] * (1.0f / (NN * CC)) - mu * mu;
  float rs = rsqrtf(var + EPSV);
  for (int i = tid; i < NN * CC; i += 256) {
    int n = i / CC, c = i % CC;
    float v = (xp[i] - mu) * rs * g[i] + beta[i];
    out[(((long)b * CC + c) * T2 + t2) * NN + n] = v;
  }
}

// ---------------------------------------------------------------------------
extern "C" void kernel_launch(void* const* d_in, const int* in_sizes, int n_in,
                              void* d_out, int out_size, void* d_ws, size_t ws_size,
                              hipStream_t stream) {
  (void)in_sizes; (void)n_in; (void)out_size; (void)ws_size;
  const float* x    = (const float*)d_in[0];
  const float* gso  = (const float*)d_in[1];
  const float* c1w  = (const float*)d_in[2];
  const float* c1b  = (const float*)d_in[3];
  const float* chw  = (const float*)d_in[4];
  const float* chb  = (const float*)d_in[5];
  const float* wq   = (const float*)d_in[6];
  const float* wk   = (const float*)d_in[7];
  const float* wvp  = (const float*)d_in[8];
  const float* asrc = (const float*)d_in[9];
  const float* adst = (const float*)d_in[10];
  const float* gg   = (const float*)d_in[11];
  const float* gb   = (const float*)d_in[12];
  const float* c2w  = (const float*)d_in[13];
  const float* c2b  = (const float*)d_in[14];
  const float* lng  = (const float*)d_in[15];
  const float* lnb  = (const float*)d_in[16];
  float* out = (float*)d_out;

  char* w = (char*)d_ws;
  size_t off = 0;
  auto alloc = [&](size_t bytes) -> char* {
    char* p = w + off; off += (bytes + 255) & ~(size_t)255; return p;
  };
  float* x1f  = (float*)alloc((size_t)BT1 * NN * CC * 4);  // later reused as x_f
  float* gcf  = (float*)alloc((size_t)BT1 * NN * CC * 4);
  float* gatf = (float*)alloc((size_t)BT1 * NN * CC * 4);
  float* x2f  = (float*)alloc((size_t)BT2 * NN * CC * 4);
  float* ssrc = (float*)alloc((size_t)BT1 * NN * HH * 4);
  float* sdst = (float*)alloc((size_t)BT1 * NN * HH * 4);
  float* aq   = (float*)alloc(CC * HH * 4);
  float* ak   = (float*)alloc(CC * HH * 4);
  f16* x1h   = (f16*)alloc((size_t)BT1 * NN * CC * 2);   // A-side, node-major
  f16* x1t   = (f16*)alloc((size_t)BT1 * CC * NP * 2);   // B-side, c-major, padded
  f16* y1h   = (f16*)alloc((size_t)BT1 * NN * CC * 2);
  f16* y1t   = (f16*)alloc((size_t)BT1 * CC * NP * 2);
  f16* y2h   = (f16*)alloc((size_t)BT1 * NN * CC * 2);
  f16* vt    = (f16*)alloc((size_t)BT1 * CC * NP * 2);
  f16* xfh   = (f16*)alloc((size_t)BT1 * NN * CC * 2);
  f16* Gf    = (f16*)alloc((size_t)NN * NP * 2);
  f16* wvh   = (f16*)alloc(CC * CC * 2);
  f16* chWt  = (f16*)alloc(3 * CC * CC * 2);
  f16* c2Wt  = (f16*)alloc(3 * 128 * CC * 2);

  k_prep<<<2048, 256, 0, stream>>>(gso, wq, wk, wvp, asrc, adst, chw, c2w,
                                   Gf, wvh, chWt, c2Wt, aq, ak,
                                   x1t, y1t, vt);
  {
    long tot = (long)BT1 * NN * CC;
    k_tconv1<<<(tot + 255) / 256, 256, 0, stream>>>(x, c1w, c1b, x1f, x1h, x1t);
  }
  k_gx<<<BT1 * 25, 128, 0, stream>>>(Gf, x1t, y1h, y1t);
  k_gy<<<BT1 * 25, 128, 0, stream>>>(Gf, y1t, x1f, y2h);
  k_v<<<BT1 * 25, 128, 0, stream>>>(x1h, wvh, vt);
  {
    long tot = (long)BT1 * NN * HH;
    k_ssd<<<(tot + 255) / 256, 256, 0, stream>>>(x1f, aq, ak, ssrc, sdst);
  }
  k_cheb<<<BT1 * 25, 128, 0, stream>>>(x1h, y1h, y2h, chWt, chb, x1f, gcf);
  k_attn<<<BT1 * HH * 25, 32, 0, stream>>>(gso, ssrc, sdst, vt, gatf);
  {
    long tot = (long)BT1 * NN;
    // x1f is dead as x1 after this point; reuse it to hold x_f
    k_lncomb<<<(tot + 255) / 256, 256, 0, stream>>>(gatf, gcf, gg, gb, x1f, xfh);
  }
  k_tconv2<<<BT2 * 25, 128, 0, stream>>>(xfh, x1f, c2Wt, c2b, x2f);
  k_finln<<<BT2, 256, 0, stream>>>(x2f, lng, lnb, out);
}
